// CNNMIMOEncoder_43765716746669
// MI455X (gfx1250) — compile-verified, matrix operand order checked by reading, not run on hardware
//
#include <hip/hip_runtime.h>
#include <hip/hip_bf16.h>
#include <math.h>

typedef __attribute__((ext_vector_type(16))) __bf16 v16bf;
typedef __attribute__((ext_vector_type(8)))  float  v8f;
typedef unsigned short ushort_t;
typedef unsigned int   uint_t;
typedef __attribute__((address_space(3))) ushort_t lds_us;
typedef lds_us* lds_ptr;

#define B_SZ   4096
#define HFEAT  8192      // 2*4096 elements per batch image
#define WIDTH  4096
#define KDIM   8192      // GEMM K
#define NOUT   3072      // GEMM N
#define NV     2048
#define NU     1024
#define NBLK   8
#define BN_EPS 1e-5f

static __device__ __forceinline__ ushort_t f2bf(float f) {
    uint_t u = __float_as_uint(f);
    uint_t r = (u + 0x7FFFu + ((u >> 16) & 1u)) >> 16;   // round-to-nearest-even
    return (ushort_t)r;
}
static __device__ __forceinline__ float bf2f(uint_t lo16) {
    return __uint_as_float(lo16 << 16);
}
static __device__ __forceinline__ float elu1(float t) {
    return t > 0.f ? t : (__expf(t) - 1.f);
}

// 3x3 conv tap on a 2 x 4096 LDS image with zero 'SAME' padding
static __device__ __forceinline__ float conv9(const float (*tile)[WIDTH],
                                              const float w[3][3], float bias,
                                              int c, int j) {
    float a = bias;
#pragma unroll
    for (int dc = -1; dc <= 1; ++dc) {
        int cc = c + dc;
        if (cc < 0 || cc > 1) continue;
#pragma unroll
        for (int dj = -1; dj <= 1; ++dj) {
            int jj = j + dj;
            if (jj < 0 || jj >= WIDTH) continue;
            a += w[dc + 1][dj + 1] * tile[cc][jj];
        }
    }
    return a;
}

// ---------------------------------------------------------------- stats init
__global__ void k_init_stats(float* stats) {
    if (threadIdx.x < 32) stats[threadIdx.x] = 0.f;
}

// ---------------------------------------------------------- W -> bf16 (pairs)
__global__ void k_wconv(const float* __restrict__ W, uint_t* __restrict__ WB, int n2) {
    int i = blockIdx.x * blockDim.x + threadIdx.x;
    if (i >= n2) return;
    float2 f = ((const float2*)W)[i];
    WB[i] = (uint_t)f2bf(f.x) | ((uint_t)f2bf(f.y) << 16);
}

// -------------------------------------------------- conv block 0 (rearrange)
__global__ void __launch_bounds__(256)
k_conv_first(const float* __restrict__ x, const float* __restrict__ cw,
             const float* __restrict__ cb, uint_t* __restrict__ out,
             float* __restrict__ stats) {
    __shared__ float tile[2][WIDTH];
    __shared__ float r1[256], r2[256];
    const int b = blockIdx.x, tid = threadIdx.x;

    // coalesced over x-order, scatter into LDS:
    // h[c, (nrx*32+ntx)*8+user] = x[b, user, nrx, ntx, c]
    for (int xi = tid; xi < HFEAT; xi += 256) {
        int c   = xi & 1;
        int ntx = (xi >> 1) & 31;
        int nrx = (xi >> 6) & 15;
        int u   = xi >> 10;
        int j   = (nrx * 32 + ntx) * 8 + u;
        tile[c][j] = x[(size_t)b * HFEAT + xi];
    }
    __syncthreads();

    float w[3][3];
#pragma unroll
    for (int kh = 0; kh < 3; ++kh)
#pragma unroll
        for (int kw = 0; kw < 3; ++kw) w[kh][kw] = cw[kh * 3 + kw];
    const float bias = cb[0];

    float ls = 0.f, lss = 0.f;
    for (int i2 = tid; i2 < HFEAT / 2; i2 += 256) {
        int i = i2 * 2;
        int c = i >> 12, j = i & 4095;
        float a0 = conv9(tile, w, bias, c, j);
        float a1 = conv9(tile, w, bias, c, j + 1);
        out[(size_t)b * (HFEAT / 2) + i2] = (uint_t)f2bf(a0) | ((uint_t)f2bf(a1) << 16);
        ls += a0 + a1; lss += a0 * a0 + a1 * a1;
    }
    r1[tid] = ls; r2[tid] = lss;
    __syncthreads();
    for (int s = 128; s > 0; s >>= 1) {
        if (tid < s) { r1[tid] += r1[tid + s]; r2[tid] += r2[tid + s]; }
        __syncthreads();
    }
    if (tid == 0) { atomicAdd(&stats[0], r1[0]); atomicAdd(&stats[8], r2[0]); }
}

// ------------- conv block i (bf16 in/out): load applies BN(i-1)+ELU on the fly
__global__ void __launch_bounds__(256)
k_conv_mid(const uint_t* __restrict__ in, const float* __restrict__ cw,
           const float* __restrict__ cb, uint_t* __restrict__ out,
           float* __restrict__ stats, int blk) {
    __shared__ float tile[2][WIDTH];
    __shared__ float r1[256], r2[256];
    const int b = blockIdx.x, tid = threadIdx.x;
    const float sc = stats[16 + 2 * (blk - 1)];
    const float sh = stats[17 + 2 * (blk - 1)];

    for (int i2 = tid; i2 < HFEAT / 2; i2 += 256) {
        uint_t p = in[(size_t)b * (HFEAT / 2) + i2];
        int i = i2 * 2;
        int c = i >> 12, j = i & 4095;
        tile[c][j]     = elu1(sc * bf2f(p & 0xFFFFu) + sh);
        tile[c][j + 1] = elu1(sc * bf2f(p >> 16) + sh);
    }
    __syncthreads();

    float w[3][3];
#pragma unroll
    for (int kh = 0; kh < 3; ++kh)
#pragma unroll
        for (int kw = 0; kw < 3; ++kw) w[kh][kw] = cw[blk * 9 + kh * 3 + kw];
    const float bias = cb[blk];

    float ls = 0.f, lss = 0.f;
    for (int i2 = tid; i2 < HFEAT / 2; i2 += 256) {
        int i = i2 * 2;
        int c = i >> 12, j = i & 4095;
        float a0 = conv9(tile, w, bias, c, j);
        float a1 = conv9(tile, w, bias, c, j + 1);
        out[(size_t)b * (HFEAT / 2) + i2] = (uint_t)f2bf(a0) | ((uint_t)f2bf(a1) << 16);
        ls += a0 + a1; lss += a0 * a0 + a1 * a1;
    }
    r1[tid] = ls; r2[tid] = lss;
    __syncthreads();
    for (int s = 128; s > 0; s >>= 1) {
        if (tid < s) { r1[tid] += r1[tid + s]; r2[tid] += r2[tid + s]; }
        __syncthreads();
    }
    if (tid == 0) { atomicAdd(&stats[blk], r1[0]); atomicAdd(&stats[8 + blk], r2[0]); }
}

// --------------------------------------------------- finalize BN stats per block
__global__ void k_stats(float* stats, const float* __restrict__ g,
                        const float* __restrict__ bt, int blk) {
    if (threadIdx.x == 0) {
        const float N = 33554432.0f;   // B_SZ * HFEAT
        float mean = stats[blk] / N;
        float var  = stats[8 + blk] / N - mean * mean;
        float inv  = rsqrtf(var + BN_EPS);
        float sc   = g[blk] * inv;
        stats[16 + 2 * blk] = sc;
        stats[17 + 2 * blk] = bt[blk] - sc * mean;
    }
}

// -------------------------------- final BN(7)+ELU -> bf16 GEMM A matrix (pairs)
__global__ void k_actfin(const uint_t* __restrict__ H, const float* __restrict__ stats,
                         uint_t* __restrict__ A, int n2) {
    int i = blockIdx.x * blockDim.x + threadIdx.x;
    if (i >= n2) return;
    float sc = stats[16 + 2 * 7], sh = stats[17 + 2 * 7];
    uint_t p = H[i];
    float e0 = elu1(sc * bf2f(p & 0xFFFFu) + sh);
    float e1 = elu1(sc * bf2f(p >> 16) + sh);
    A[i] = (uint_t)f2bf(e0) | ((uint_t)f2bf(e1) << 16);
}

// ---------------------------------------------------------------- bf16 WMMA GEMM
// C[4096x3072] = A[4096x8192] * W^T (W row-major [3072x8192] => B-tile = W rows).
// 128x128 C-tile / 256 threads; 8 waves each own 32x64 (2x4 WMMA tiles), K step 64.
// Tiles staged into LDS with CDNA5 async global->LDS loads (ASYNCcnt).
// B fragments are loaded one-at-a-time and consumed immediately to keep peak
// fragment liveness at 3 frags (24 VGPRs) so accumulators never get rotated.
__global__ void __launch_bounds__(256)
k_gemm(const ushort_t* __restrict__ A, const ushort_t* __restrict__ Bw,
       const float* __restrict__ bias, float* __restrict__ Y) {
    __shared__ __align__(16) ushort_t As[128 * 64];
    __shared__ __align__(16) ushort_t Bs[128 * 64];

    const int tid  = threadIdx.x;
    const int lane = tid & 31;
    const int wv   = tid >> 5;            // wave 0..7
    const int wm   = (wv >> 1) * 32;      // wave M origin within tile
    const int wn   = (wv & 1) * 64;       // wave N origin within tile
    const int m0   = blockIdx.y * 128;
    const int n0   = blockIdx.x * 128;
    const int lr   = tid >> 1;            // 0..127: tile row this thread copies
    const int lh   = (tid & 1) * 32;      // 0/32: which 32-element (64B) K half
    const int mrow = lane & 15;
    const int kbase = (lane < 16) ? 0 : 8;  // ISA 16-bit A-layout K base

    const v8f vzero = {0.f, 0.f, 0.f, 0.f, 0.f, 0.f, 0.f, 0.f};
    v8f acc0[4], acc1[4];
#pragma unroll
    for (int j = 0; j < 4; ++j) { acc0[j] = vzero; acc1[j] = vzero; }

    const ushort_t* aSrc = A  + (size_t)(m0 + lr) * KDIM + lh;
    const ushort_t* bSrc = Bw + (size_t)(n0 + lr) * KDIM + lh;
    lds_ptr aLp = (lds_ptr)(As + lr * 64 + lh);   // raw LDS offset in one VGPR
    lds_ptr bLp = (lds_ptr)(Bs + lr * 64 + lh);

    for (int kt = 0; kt < KDIM / 64; ++kt, aSrc += 64, bSrc += 64) {
        __syncthreads();
        // each thread copies 64B of A and 64B of W: async DMA into LDS,
        // same IOFFSET applies to both global and LDS sides (ISA 15.18.3)
        asm volatile(
            "global_load_async_to_lds_b128 %0, %2, off\n\t"
            "global_load_async_to_lds_b128 %0, %2, off offset:16\n\t"
            "global_load_async_to_lds_b128 %0, %2, off offset:32\n\t"
            "global_load_async_to_lds_b128 %0, %2, off offset:48\n\t"
            "global_load_async_to_lds_b128 %1, %3, off\n\t"
            "global_load_async_to_lds_b128 %1, %3, off offset:16\n\t"
            "global_load_async_to_lds_b128 %1, %3, off offset:32\n\t"
            "global_load_async_to_lds_b128 %1, %3, off offset:48\n\t"
            :: "v"(aLp), "v"(bLp), "v"(aSrc), "v"(bSrc) : "memory");
        if (kt + 1 < KDIM / 64) {             // uniform branch
            __builtin_prefetch(aSrc + 64, 0, 1);
            __builtin_prefetch(bSrc + 64, 0, 1);
        }
        asm volatile("s_wait_asynccnt 0x0" ::: "memory");
        __syncthreads();

#pragma unroll
        for (int ks = 0; ks < 2; ++ks) {
            union F { v16bf v; uint_t u[8]; };
            F fa0, fa1;
            {
                const ushort_t* p0 = As + (wm + mrow) * 64 + ks * 32 + kbase;
                const ushort_t* p1 = As + (wm + 16 + mrow) * 64 + ks * 32 + kbase;
                ((uint4*)fa0.u)[0] = ((const uint4*)p0)[0];        // K base..+7
                ((uint4*)fa0.u)[1] = ((const uint4*)(p0 + 16))[0]; // K base+16..+23
                ((uint4*)fa1.u)[0] = ((const uint4*)p1)[0];
                ((uint4*)fa1.u)[1] = ((const uint4*)(p1 + 16))[0];
            }
#pragma unroll
            for (int in = 0; in < 4; ++in) {
                F fb;
                const ushort_t* pb = Bs + (wn + in * 16 + mrow) * 64 + ks * 32 + kbase;
                ((uint4*)fb.u)[0] = ((const uint4*)pb)[0];
                ((uint4*)fb.u)[1] = ((const uint4*)(pb + 16))[0];
                acc0[in] = __builtin_amdgcn_wmma_f32_16x16x32_bf16(
                    false, fa0.v, false, fb.v, (short)0, acc0[in], false, false);
                acc1[in] = __builtin_amdgcn_wmma_f32_16x16x32_bf16(
                    false, fa1.v, false, fb.v, (short)0, acc1[in], false, false);
            }
        }
    }

    // C layout: VGPR r -> M = r (+8 for lanes 16-31), N = lane%16
    const int mlane = (lane < 16) ? 0 : 8;
    const int nlane = lane & 15;
#pragma unroll
    for (int in = 0; in < 4; ++in) {
        int Nc = n0 + wn + in * 16 + nlane;
        float bl = bias[Nc];
        int Mb0 = m0 + wm + mlane;
#pragma unroll
        for (int r = 0; r < 8; ++r)
            Y[(size_t)(Mb0 + r) * NOUT + Nc] = acc0[in][r] + bl;
#pragma unroll
        for (int r = 0; r < 8; ++r)
            Y[(size_t)(Mb0 + 16 + r) * NOUT + Nc] = acc1[in][r] + bl;
    }
}

// ---------------------------------------------- per-row v-normalize + split u/v
__global__ void __launch_bounds__(256)
k_outnorm(const float* __restrict__ Y, float* __restrict__ out) {
    __shared__ float red[256];
    const int b = blockIdx.x, tid = threadIdx.x;
    const float* row = Y + (size_t)b * NOUT;
    float s = 0.f;
    for (int n = tid; n < NV; n += 256) { float v = row[n]; s += v * v; }
    red[tid] = s;
    __syncthreads();
    for (int st = 128; st > 0; st >>= 1) {
        if (tid < st) red[tid] += red[tid + st];
        __syncthreads();
    }
    const float inv = 1.f / sqrtf(red[0]);
    float* uo = out;                          // u: [B,1024] first in tuple
    float* vo = out + (size_t)B_SZ * NU;      // v: [B,2048]
    for (int n = tid; n < NOUT; n += 256) {
        float v = row[n];
        if (n < NV) vo[(size_t)b * NV + n] = v * inv;
        else        uo[(size_t)b * NU + (n - NV)] = v;
    }
}

// ================================================================= launcher
extern "C" void kernel_launch(void* const* d_in, const int* in_sizes, int n_in,
                              void* d_out, int out_size, void* d_ws, size_t ws_size,
                              hipStream_t stream) {
    const float* x    = (const float*)d_in[0];   // (4096,8,16,32,2)
    const float* cw   = (const float*)d_in[1];   // (8,1,1,3,3)
    const float* cb   = (const float*)d_in[2];   // (8,1)
    const float* bn_g = (const float*)d_in[3];   // (8,1)
    const float* bn_b = (const float*)d_in[4];   // (8,1)
    const float* W    = (const float*)d_in[5];   // (3072,8192)
    const float* blin = (const float*)d_in[6];   // (3072,)

    const size_t SZ_HB = (size_t)B_SZ * HFEAT * sizeof(ushort_t);   // 67 MB (bf16)
    const size_t SZ_WB = (size_t)NOUT * KDIM * sizeof(ushort_t);    // 48 MB

    char* ws = (char*)d_ws;
    uint_t*   Hb0   = (uint_t*)ws;                       // bf16 pairs
    uint_t*   Hb1   = (uint_t*)(ws + SZ_HB);
    ushort_t* WB    = (ushort_t*)(ws + 2 * SZ_HB);
    float*    stats = (float*)(ws + 2 * SZ_HB + SZ_WB);  // sums[8], ssq[8], params[16]
    ushort_t* AB    = (ushort_t*)Hb0;   // bf16 A overwrites Hb0 (raw6, dead)
    float*    Y     = (float*)Hb1;      // y (50MB) overwrites Hb1 after actfin

    // 0) zero stat accumulators (graph-replay safe) + convert W to bf16
    k_init_stats<<<1, 32, 0, stream>>>(stats);
    {
        int n2 = NOUT * KDIM / 2;
        k_wconv<<<n2 / 256, 256, 0, stream>>>(W, (uint_t*)WB, n2);
    }

    // 1) conv chain; bf16 ping-pong buffers (134MB total) stay L2-resident
    k_conv_first<<<B_SZ, 256, 0, stream>>>(x, cw, cb, Hb0, stats);
    k_stats<<<1, 32, 0, stream>>>(stats, bn_g, bn_b, 0);
    for (int i = 1; i < NBLK; ++i) {
        const uint_t* src = (i & 1) ? Hb0 : Hb1;
        uint_t*       dst = (i & 1) ? Hb1 : Hb0;
        k_conv_mid<<<B_SZ, 256, 0, stream>>>(src, cw, cb, dst, stats, i);
        k_stats<<<1, 32, 0, stream>>>(stats, bn_g, bn_b, i);
    }
    // chain ends with raw block-7 output in Hb1

    // 2) BN(7)+ELU -> bf16 A
    {
        int n2 = B_SZ * HFEAT / 2;
        k_actfin<<<n2 / 256, 256, 0, stream>>>(Hb1, stats, (uint_t*)AB, n2);
    }

    // 3) WMMA GEMM: Y = A * W^T + b
    {
        dim3 grid(NOUT / 128, B_SZ / 128);   // (24, 32)
        k_gemm<<<grid, 256, 0, stream>>>(AB, WB, blin, Y);
    }

    // 4) per-row v normalization, write (u, v) to d_out
    k_outnorm<<<B_SZ, 256, 0, stream>>>(Y, (float*)d_out);
}
